// InfoNCE_1056561955228
// MI455X (gfx1250) — matrix operand order repaired and checked by reading, not compiled, not run on hardware
//
#include <hip/hip_runtime.h>
#include <math.h>

// Problem constants (fixed by the reference).
#define B_ 4
#define S_ 4096
#define N_ 256
#define CBLK 64                 // columns handled per workgroup
#define HALF_TILES 128          // s-tiles (of 16 rows) per s-half: S/16/2
#define LDS_PAD 8               // bf16 elements of per-row padding
#define LDS_STRIDE (N_ + LDS_PAD)

typedef __attribute__((ext_vector_type(16))) __bf16 v16bf;
typedef __attribute__((ext_vector_type(8)))  float  v8f;
typedef __attribute__((ext_vector_type(4)))  unsigned int v4u;
typedef __attribute__((ext_vector_type(8)))  int v8i;
typedef __attribute__((ext_vector_type(4)))  int v4i;

#if __has_builtin(__builtin_amdgcn_tensor_load_to_lds)
#define HAVE_TDM 1
#else
#define HAVE_TDM 0
#endif

union BFV {
  v16bf v;
  uint4 q[2];
  unsigned short us[16];
};

// round-to-nearest-even f32 -> bf16 (bit trick)
__device__ __forceinline__ unsigned short bf16_rne(float x) {
  unsigned u = __float_as_uint(x);
  unsigned r = u + 0x7FFFu + ((u >> 16) & 1u);
  return (unsigned short)(r >> 16);
}
__device__ __forceinline__ float bf16_to_f32(unsigned short h) {
  return __uint_as_float(((unsigned)h) << 16);
}

__device__ __forceinline__ void wait_tensorcnt0() {
#if __has_builtin(__builtin_amdgcn_s_wait_tensorcnt)
  __builtin_amdgcn_s_wait_tensorcnt(0);
#else
  asm volatile("s_wait_tensorcnt 0x0" ::: "memory");
#endif
}

__global__ void infonce_zero_kernel(float* out) {
  if (threadIdx.x == 0) out[0] = 0.0f;
}

// ============ prep: split f32 -> (bf16 hi, bf16 lo) global planes ==========
// Memory bound (~67MB total traffic); conversion VALU is hidden behind HBM.
__global__ void __launch_bounds__(256)
cvt_split_kernel(const float* __restrict__ src,
                 unsigned short* __restrict__ hi,
                 unsigned short* __restrict__ lo) {
  const int i = (blockIdx.x * 256 + threadIdx.x) * 4;
  float4 x = *(const float4*)(src + i);
  float xv[4] = {x.x, x.y, x.z, x.w};
  unsigned hp[2], lp[2];
#pragma unroll
  for (int p = 0; p < 2; ++p) {
    unsigned short h0 = bf16_rne(xv[2 * p]);
    unsigned short h1 = bf16_rne(xv[2 * p + 1]);
    unsigned short l0 = bf16_rne(xv[2 * p] - bf16_to_f32(h0));
    unsigned short l1 = bf16_rne(xv[2 * p + 1] - bf16_to_f32(h1));
    hp[p] = (unsigned)h0 | ((unsigned)h1 << 16);
    lp[p] = (unsigned)l0 | ((unsigned)l1 << 16);
  }
  *(uint2*)(hi + i) = make_uint2(hp[0], hp[1]);
  *(uint2*)(lo + i) = make_uint2(lp[0], lp[1]);
}

#if HAVE_TDM
// One TDM descriptor moves a 16x256 bf16 tile (row-major, stride 256 elems)
// from global into LDS, inserting 16B of padding after every 512B row so the
// LDS layout has stride LDS_STRIDE without any shader-side address math.
// This toolchain's builtin takes 6 args:
//   (uint32x4 g0, int32x8 g1, int32x4 g2, int32x4 g3, int32x8 extra, i32 cpol)
__device__ __forceinline__ void tdm_load_tile(const unsigned short* gsrc,
                                              unsigned lds_off) {
  unsigned long long ga = (unsigned long long)(uintptr_t)gsrc;
  v4u g0 = {1u,                                   // count=1 (valid descriptor)
            lds_off,                              // lds_addr (bytes)
            (unsigned)(ga & 0xFFFFFFFFu),         // global_addr[31:0]
            (unsigned)((ga >> 32) & 0x01FFFFFFu) | (2u << 30)}; // ga[56:32]|type=2
  v8i g1 = {(int)((1u << 16)      // data_size = 2 bytes
                  | (1u << 20)    // pad_enable
                  | (6u << 22)    // pad_interval: 128 DWORDs (=512B, one row)
                  | (3u << 25)),  // pad_amount: 4 DWORDs (=16B = LDS_PAD bf16)
            (int)(256u << 16),    // tensor_dim0 = 256 (elems)
            (int)(4096u << 16),   // tensor_dim1 = 4096 (rows)
            (int)(256u << 16),    // tile_dim0 = 256
            16,                   // tile_dim1 = 16 rows
            256,                  // tensor_dim0_stride = 256 elems
            0, 0};
  v4i z4 = {0, 0, 0, 0};
  v8i z8 = {0, 0, 0, 0, 0, 0, 0, 0};
  __builtin_amdgcn_tensor_load_to_lds(g0, g1, z4, z4, z8, 0);
}
#endif

// ============ main fused kernel: bf16-hi/lo planes already in d_ws ==========
__global__ void __launch_bounds__(256)
infonce_tdm_kernel(const unsigned short* __restrict__ z1hi,
                   const unsigned short* __restrict__ z1lo,
                   const unsigned short* __restrict__ z2hi,
                   const unsigned short* __restrict__ z2lo,
                   float* __restrict__ out) {
  // double-buffered z1 s-tiles: [buf][half][row][k]
  __shared__ unsigned short Ahi[2][2][16][LDS_STRIDE];
  __shared__ unsigned short Alo[2][2][16][LDS_STRIDE];
  __shared__ float2 mlred[2][4][32];
  __shared__ float  diagv[CBLK];

  const int tid = threadIdx.x;
  const int b   = blockIdx.x >> 6;
  const int cb  = blockIdx.x & 63;
  const size_t base = (size_t)b * S_ * N_;

  const int lane = tid & 31;
  const int w    = tid >> 5;
  const int tile = w & 3;
  const int chf  = w >> 2;
  const int jhi  = lane >> 4;
  const int nlo  = lane & 15;
  const int t0   = cb * CBLK + tile * 16;
  const int dtile = t0 >> 4;

  // ---- B operand (z2 tile) pinned in VGPRs, straight bf16 loads, no cvt ----
  v16bf Bhi[8], Blo[8];
  {
    const size_t roff = base + (size_t)(t0 + nlo) * N_;
#pragma unroll
    for (int kk = 0; kk < 8; ++kk) {
      const int off = kk * 32 + (jhi ? 16 : 0);
      BFV hb, lb;
      hb.q[0] = *(const uint4*)(z2hi + roff + off);
      hb.q[1] = *(const uint4*)(z2hi + roff + off + 8);
      lb.q[0] = *(const uint4*)(z2lo + roff + off);
      lb.q[1] = *(const uint4*)(z2lo + roff + off + 8);
      Bhi[kk] = hb.v;
      Blo[kk] = lb.v;
    }
  }

#if !HAVE_TDM
  // fallback staging role: 256 threads copy 2 tiles/iter (bf16, no conversion)
  const int srr = tid >> 3;
  const int sh  = srr >> 4;
  const int sr  = srr & 15;
  const int skb = (tid & 7) * 32;
  uint4 ph[4], pl[4];
  auto stage_load = [&](int it) {
    const size_t g = base + (size_t)((sh * HALF_TILES + it) * 16 + sr) * N_ + skb;
    const uint4* gh = (const uint4*)(z1hi + g);
    const uint4* gl = (const uint4*)(z1lo + g);
#pragma unroll
    for (int i = 0; i < 4; ++i) { ph[i] = gh[i]; pl[i] = gl[i]; }
  };
  auto stage_store = [&](int buf) {
    uint4* dh = (uint4*)&Ahi[buf][sh][sr][skb];
    uint4* dl = (uint4*)&Alo[buf][sh][sr][skb];
#pragma unroll
    for (int i = 0; i < 4; ++i) { dh[i] = ph[i]; dl[i] = pl[i]; }
  };
#endif

  // ---- prologue: stage s-tile 0 of both halves into buffer 0 ----
#if HAVE_TDM
  if (w == 0) {
#pragma unroll
    for (int h = 0; h < 2; ++h) {
      const size_t g = base + (size_t)(h * HALF_TILES) * 16 * N_;
      tdm_load_tile(z1hi + g, (unsigned)(uintptr_t)&Ahi[0][h][0][0]);
      tdm_load_tile(z1lo + g, (unsigned)(uintptr_t)&Alo[0][h][0][0]);
    }
    wait_tensorcnt0();
  }
#else
  stage_load(0);
  stage_store(0);
#endif
  __syncthreads();

  // ---- main loop over 128 s-tiles per half ----
  float m = -INFINITY, l = 0.0f;

  for (int it = 0; it < HALF_TILES; ++it) {
    const int buf = it & 1, nbuf = buf ^ 1;
    const bool more = (it + 1) < HALF_TILES;

#if HAVE_TDM
    if (w == 0 && more) {    // DMA tile it+1 while we compute tile it
#pragma unroll
      for (int h = 0; h < 2; ++h) {
        const size_t g = base + (size_t)((h * HALF_TILES + it + 1) * 16) * N_;
        tdm_load_tile(z1hi + g, (unsigned)(uintptr_t)&Ahi[nbuf][h][0][0]);
        tdm_load_tile(z1lo + g, (unsigned)(uintptr_t)&Alo[nbuf][h][0][0]);
      }
    }
#else
    if (more) stage_load(it + 1);
#endif

    // 16x16 logits tile, hi/lo-split bf16: three INDEPENDENT accumulator
    // chains (hi*hi, hi*lo, lo*hi) so consecutive WMMAs can pipeline.
    v8f a0 = {0.f, 0.f, 0.f, 0.f, 0.f, 0.f, 0.f, 0.f};
    v8f a1 = a0, a2 = a0;
#pragma unroll
    for (int kk = 0; kk < 8; ++kk) {
      const int c0 = kk * 32 + (jhi ? 16 : 0);
      BFV ah, al;
      ah.q[0] = *(const uint4*)&Ahi[buf][chf][nlo][c0];
      ah.q[1] = *(const uint4*)&Ahi[buf][chf][nlo][c0 + 8];
      al.q[0] = *(const uint4*)&Alo[buf][chf][nlo][c0];
      al.q[1] = *(const uint4*)&Alo[buf][chf][nlo][c0 + 8];
      a0 = __builtin_amdgcn_wmma_f32_16x16x32_bf16(
          false, ah.v, false, Bhi[kk], (short)0, a0, false, false);
      a1 = __builtin_amdgcn_wmma_f32_16x16x32_bf16(
          false, ah.v, false, Blo[kk], (short)0, a1, false, false);
      a2 = __builtin_amdgcn_wmma_f32_16x16x32_bf16(
          false, al.v, false, Bhi[kk], (short)0, a2, false, false);
    }
    v8f acc = a0 + a1 + a2;

    // online logsumexp update (per lane: 8 rows of one column)
    float tm = acc[0];
#pragma unroll
    for (int v = 1; v < 8; ++v) tm = fmaxf(tm, acc[v]);
    const float nm = fmaxf(m, tm);
    const float sc = __expf(m - nm);
    float ssum = 0.0f;
#pragma unroll
    for (int v = 0; v < 8; ++v) ssum += __expf(acc[v] - nm);
    l = l * sc + ssum;
    m = nm;

    // diagonal capture (C layout: col n in lanes {n, n+16}, row = vgpr idx)
    const int gtile = chf * HALF_TILES + it;
    if (gtile == dtile) {
      float dv = acc[0];
      const int want = lane & 7;
#pragma unroll
      for (int v = 1; v < 8; ++v) dv = (want == v) ? acc[v] : dv;
      if ((lane < 8) || (lane >= 24)) diagv[tile * 16 + nlo] = dv;
    }

#if HAVE_TDM
    if (w == 0 && more) wait_tensorcnt0();
#else
    if (more) stage_store(nbuf);
#endif
    __syncthreads();
  }

  // ---- combine 4 partials per column, accumulate loss ----
  mlred[chf][tile][lane] = make_float2(m, l);
  __syncthreads();

  if (tid < CBLK) {
    const int tl = tid >> 4, nn = tid & 15;
    float2 p0 = mlred[0][tl][nn];
    float2 p1 = mlred[0][tl][nn + 16];
    float2 p2 = mlred[1][tl][nn];
    float2 p3 = mlred[1][tl][nn + 16];
    float M = fmaxf(fmaxf(p0.x, p1.x), fmaxf(p2.x, p3.x));
    float L = p0.y * __expf(p0.x - M) + p1.y * __expf(p1.x - M) +
              p2.y * __expf(p2.x - M) + p3.y * __expf(p3.x - M);
    float lse = M + __logf(L);
    float contrib = (lse - diagv[tid]) * (1.0f / (float)(B_ * S_));
    atomicAdd(out, contrib);
  }
}

// ============ round-1 style fallback (no workspace needed) =================
__global__ void __launch_bounds__(256)
infonce_inline_kernel(const float* __restrict__ z1,
                      const float* __restrict__ z2,
                      float* __restrict__ out) {
  __shared__ unsigned short Ahi[2][16][LDS_STRIDE];
  __shared__ unsigned short Alo[2][16][LDS_STRIDE];
  __shared__ float2 mlred[2][4][32];
  __shared__ float  diagv[CBLK];

  const int tid = threadIdx.x;
  const int b   = blockIdx.x >> 6;
  const int cb  = blockIdx.x & 63;
  const float* z1b = z1 + (size_t)b * S_ * N_;
  const float* z2b = z2 + (size_t)b * S_ * N_;
  const int lane = tid & 31;
  const int w    = tid >> 5;
  const int tile = w & 3;
  const int chf  = w >> 2;
  const int jhi  = lane >> 4;
  const int nlo  = lane & 15;
  const int t0   = cb * CBLK + tile * 16;
  const int dtile = t0 >> 4;
  const int srr = tid >> 3, sh = srr >> 4, sr = srr & 15, skb = (tid & 7) * 32;

  v16bf Bhi[8], Blo[8];
  {
    const float* z2row = z2b + (size_t)(t0 + nlo) * N_;
#pragma unroll
    for (int kk = 0; kk < 8; ++kk) {
      const float4* s4 = (const float4*)(z2row + kk * 32 + (jhi ? 16 : 0));
      float4 c0 = s4[0], c1 = s4[1], c2 = s4[2], c3 = s4[3];
      float xs[16] = {c0.x, c0.y, c0.z, c0.w, c1.x, c1.y, c1.z, c1.w,
                      c2.x, c2.y, c2.z, c2.w, c3.x, c3.y, c3.z, c3.w};
      BFV hb, lb;
#pragma unroll
      for (int e = 0; e < 16; ++e) {
        unsigned short hh = bf16_rne(xs[e]);
        hb.us[e] = hh;
        lb.us[e] = bf16_rne(xs[e] - bf16_to_f32(hh));
      }
      Bhi[kk] = hb.v; Blo[kk] = lb.v;
    }
  }

  float4 pf[8];
  auto stage_ptr = [&](int it) -> const float4* {
    return (const float4*)(z1b + (size_t)((sh * HALF_TILES + it) * 16 + sr) * N_ + skb);
  };
  auto load_pf = [&](int it) {
    const float4* p = stage_ptr(it);
#pragma unroll
    for (int i = 0; i < 8; ++i) pf[i] = p[i];
  };
  auto cvt_store = [&]() {
    unsigned short hs[32], ls[32];
#pragma unroll
    for (int i = 0; i < 8; ++i) {
      float xv[4] = {pf[i].x, pf[i].y, pf[i].z, pf[i].w};
#pragma unroll
      for (int e = 0; e < 4; ++e) {
        unsigned short hh = bf16_rne(xv[e]);
        hs[i * 4 + e] = hh;
        ls[i * 4 + e] = bf16_rne(xv[e] - bf16_to_f32(hh));
      }
    }
    uint4* dh = (uint4*)&Ahi[sh][sr][skb];
    uint4* dl = (uint4*)&Alo[sh][sr][skb];
#pragma unroll
    for (int i = 0; i < 4; ++i) {
      const unsigned short* hq = hs + i * 8;
      const unsigned short* lq = ls + i * 8;
      dh[i] = make_uint4((unsigned)hq[0] | ((unsigned)hq[1] << 16),
                         (unsigned)hq[2] | ((unsigned)hq[3] << 16),
                         (unsigned)hq[4] | ((unsigned)hq[5] << 16),
                         (unsigned)hq[6] | ((unsigned)hq[7] << 16));
      dl[i] = make_uint4((unsigned)lq[0] | ((unsigned)lq[1] << 16),
                         (unsigned)lq[2] | ((unsigned)lq[3] << 16),
                         (unsigned)lq[4] | ((unsigned)lq[5] << 16),
                         (unsigned)lq[6] | ((unsigned)lq[7] << 16));
    }
  };

  load_pf(0); cvt_store();
  __syncthreads();

  float m = -INFINITY, l = 0.0f;
  for (int it = 0; it < HALF_TILES; ++it) {
    const bool more = (it + 1) < HALF_TILES;
    if (more) load_pf(it + 1);
    if (it + 2 < HALF_TILES)
      __builtin_prefetch((const void*)stage_ptr(it + 2), 0, 0);

    v8f a0 = {0.f, 0.f, 0.f, 0.f, 0.f, 0.f, 0.f, 0.f};
    v8f a1 = a0, a2 = a0;
#pragma unroll
    for (int kk = 0; kk < 8; ++kk) {
      const int c0 = kk * 32 + (jhi ? 16 : 0);
      BFV ah, al;
      ah.q[0] = *(const uint4*)&Ahi[chf][nlo][c0];
      ah.q[1] = *(const uint4*)&Ahi[chf][nlo][c0 + 8];
      al.q[0] = *(const uint4*)&Alo[chf][nlo][c0];
      al.q[1] = *(const uint4*)&Alo[chf][nlo][c0 + 8];
      a0 = __builtin_amdgcn_wmma_f32_16x16x32_bf16(
          false, ah.v, false, Bhi[kk], (short)0, a0, false, false);
      a1 = __builtin_amdgcn_wmma_f32_16x16x32_bf16(
          false, ah.v, false, Blo[kk], (short)0, a1, false, false);
      a2 = __builtin_amdgcn_wmma_f32_16x16x32_bf16(
          false, al.v, false, Bhi[kk], (short)0, a2, false, false);
    }
    v8f acc = a0 + a1 + a2;

    float tm = acc[0];
#pragma unroll
    for (int v = 1; v < 8; ++v) tm = fmaxf(tm, acc[v]);
    const float nm = fmaxf(m, tm);
    const float sc = __expf(m - nm);
    float ssum = 0.0f;
#pragma unroll
    for (int v = 0; v < 8; ++v) ssum += __expf(acc[v] - nm);
    l = l * sc + ssum; m = nm;

    const int gtile = chf * HALF_TILES + it;
    if (gtile == dtile) {
      float dv = acc[0];
      const int want = lane & 7;
#pragma unroll
      for (int v = 1; v < 8; ++v) dv = (want == v) ? acc[v] : dv;
      if ((lane < 8) || (lane >= 24)) diagv[tile * 16 + nlo] = dv;
    }

    __syncthreads();
    if (more) cvt_store();
    __syncthreads();
  }

  mlred[chf][tile][lane] = make_float2(m, l);
  __syncthreads();
  if (tid < CBLK) {
    const int tl = tid >> 4, nn = tid & 15;
    float2 p0 = mlred[0][tl][nn], p1 = mlred[0][tl][nn + 16];
    float2 p2 = mlred[1][tl][nn], p3 = mlred[1][tl][nn + 16];
    float M = fmaxf(fmaxf(p0.x, p1.x), fmaxf(p2.x, p3.x));
    float L = p0.y * __expf(p0.x - M) + p1.y * __expf(p1.x - M) +
              p2.y * __expf(p2.x - M) + p3.y * __expf(p3.x - M);
    float lse = M + __logf(L);
    atomicAdd(out, (lse - diagv[tid]) * (1.0f / (float)(B_ * S_)));
  }
}

extern "C" void kernel_launch(void* const* d_in, const int* in_sizes, int n_in,
                              void* d_out, int out_size, void* d_ws, size_t ws_size,
                              hipStream_t stream) {
  (void)in_sizes; (void)n_in; (void)out_size;
  const float* z1 = (const float*)d_in[0];
  const float* z2 = (const float*)d_in[1];
  float* out = (float*)d_out;

  const size_t nelem = (size_t)B_ * S_ * N_;
  const size_t need = 4 * nelem * sizeof(unsigned short);

  infonce_zero_kernel<<<dim3(1), dim3(1), 0, stream>>>(out);

  if (ws_size >= need) {
    unsigned short* z1hi = (unsigned short*)d_ws;
    unsigned short* z1lo = z1hi + nelem;
    unsigned short* z2hi = z1lo + nelem;
    unsigned short* z2lo = z2hi + nelem;
    const int blocks = (int)(nelem / (256 * 4));   // 4 floats per thread
    cvt_split_kernel<<<dim3(blocks), dim3(256), 0, stream>>>(z1, z1hi, z1lo);
    cvt_split_kernel<<<dim3(blocks), dim3(256), 0, stream>>>(z2, z2hi, z2lo);
    infonce_tdm_kernel<<<dim3(B_ * (S_ / CBLK)), dim3(256), 0, stream>>>(
        z1hi, z1lo, z2hi, z2lo, out);
  } else {
    infonce_inline_kernel<<<dim3(B_ * (S_ / CBLK)), dim3(256), 0, stream>>>(
        z1, z2, out);
  }
}
